// ScaledDotProductAttention_41188736368659
// MI455X (gfx1250) — matrix operand order repaired and checked by reading, not compiled, not run on hardware
//
#include <hip/hip_runtime.h>
#include <stdint.h>

typedef __attribute__((ext_vector_type(16))) _Float16 v16h;
typedef __attribute__((ext_vector_type(8)))  _Float16 v8h;
typedef __attribute__((ext_vector_type(2)))  _Float16 v2h;
typedef __attribute__((ext_vector_type(8)))  float    v8f;
typedef __attribute__((ext_vector_type(4)))  float    v4f;
typedef __attribute__((ext_vector_type(4)))  int      v4i;
typedef __attribute__((ext_vector_type(4)))  unsigned int u32x4;

// Pointee types for the async-copy builtin (AS1 = global, AS3 = LDS).
typedef __attribute__((address_space(1))) v4i gv4i;
typedef __attribute__((address_space(3))) v4i lv4i;

#define S_LEN   4096
#define DMODEL  64
#define KV_TILE 64
#define NWAVES  8
#define QROWS_W 16
#define QROWS_B 128

// 1/sqrt(64) * log2(e): scores come out of QK^T already in the exp2 domain.
#define QSCALE  (0.125f * 1.44269504088896340736f)

#if __has_builtin(__builtin_amdgcn_global_load_async_to_lds_b128)
#define HAVE_ASYNC 1
#endif
#if __has_builtin(__builtin_amdgcn_permlane16)
#define HAVE_PERMLANE16 1
#endif
#if __has_builtin(__builtin_amdgcn_mov_dpp8)
#define HAVE_DPP8 1
#endif

#define DPP8SEL(a,b,c,d,e,f,g,h) \
    ((a)|((b)<<3)|((c)<<6)|((d)<<9)|((e)<<12)|((f)<<15)|((g)<<18)|((h)<<21))

// Raw hardware exp2 (single v_exp_f32 TRANS op, no denormal fixup path).
__device__ __forceinline__ float fast_exp2(float x) {
#if __has_builtin(__builtin_amdgcn_exp2f)
    return __builtin_amdgcn_exp2f(x);
#else
    return exp2f(x);
#endif
}

// xor-butterfly within each 16-lane group, kept off the DS pipe:
// xor1/2/4 via DPP8 (no old-operand tie copy), xor8 via v_permlane16_b32.
__device__ __forceinline__ float lane16_xor(float x, int step) {
#if defined(HAVE_DPP8) && defined(HAVE_PERMLANE16)
    int xi = __float_as_int(x), yi;
    switch (step) {
    case 0:  yi = __builtin_amdgcn_mov_dpp8(xi, DPP8SEL(1,0,3,2,5,4,7,6)); break;
    case 1:  yi = __builtin_amdgcn_mov_dpp8(xi, DPP8SEL(2,3,0,1,6,7,4,5)); break;
    case 2:  yi = __builtin_amdgcn_mov_dpp8(xi, DPP8SEL(4,5,6,7,0,1,2,3)); break;
    default: yi = __builtin_amdgcn_permlane16(xi, xi, 0xFEDCBA98u, 0x76543210u,
                                              false, false); break;
    }
    return __int_as_float(yi);
#elif defined(HAVE_PERMLANE16)
    const unsigned sello[4] = {0x67452301u, 0x54761032u, 0x32107654u, 0xFEDCBA98u};
    const unsigned selhi[4] = {0xEFCDAB89u, 0xDCFE98BAu, 0xBA98FEDCu, 0x76543210u};
    int xi = __float_as_int(x);
    int yi = __builtin_amdgcn_permlane16(xi, xi, sello[step], selhi[step], false, false);
    return __int_as_float(yi);
#else
    return __shfl_xor(x, 1 << step, 32);
#endif
}

// Copy 16 bytes global -> LDS. Async (ASYNCcnt-tracked) when available.
__device__ __forceinline__ void copy16(const _Float16* g, _Float16* l) {
#ifdef HAVE_ASYNC
    __builtin_amdgcn_global_load_async_to_lds_b128((gv4i*)g, (lv4i*)l, 0, 0);
#else
    *(u32x4*)l = *(const u32x4*)g;
#endif
}

__device__ __forceinline__ void wait_async() {
#ifdef HAVE_ASYNC
#if __has_builtin(__builtin_amdgcn_s_wait_asynccnt)
    __builtin_amdgcn_s_wait_asynccnt(0);
#else
    asm volatile("s_wait_asynccnt 0" ::: "memory");
#endif
#endif
}

// Permutation of kv index within a 32-block -> WMMA contraction position.
// Chosen so the P-tile can be stored to LDS as packed (g, g+16) f16 pairs and
// read back as two contiguous 32B A-operand chunks. Baked into Vt layout.
__device__ __forceinline__ int pv_perm(int s) {
    int lo = s & 15, hi = s >> 4;
    int q2 = lo >> 2, r2 = lo & 3;
    int t  = ((q2 & 1) << 3) + 2 * r2 + hi;
    int h  = q2 >> 1;
    (void)t; (void)h;
    return 2 * lo + hi;
}

// ---------------- Pre-pass 1: K f32 -> f16 row-major ----------------
__global__ __launch_bounds__(256)
void conv_k(const float* __restrict__ kin, _Float16* __restrict__ kh) {
    int idx = (blockIdx.x * 256 + threadIdx.x) * 8;
    v4f a = *(const v4f*)(kin + idx);
    v4f b = *(const v4f*)(kin + idx + 4);
    v8h hv;
    #pragma unroll
    for (int e = 0; e < 4; ++e) { hv[e] = (_Float16)a[e]; hv[4 + e] = (_Float16)b[e]; }
    *(v8h*)(kh + idx) = hv;
}

// ---------------- Pre-pass 2: V f32 [b][s][d] -> f16 Vt [b][d][perm(s)] ----------------
__global__ __launch_bounds__(256)
void conv_v(const float* __restrict__ vin, _Float16* __restrict__ vt) {
    __shared__ __align__(32) _Float16 lt[DMODEL][128];   // 16 KB tile (128 s x 64 d)
    const int tid = threadIdx.x;
    const int b   = blockIdx.x >> 5;            // 32 s-tiles of 128 per batch
    const int s0  = (blockIdx.x & 31) * 128;

    #pragma unroll
    for (int it = 0; it < 4; ++it) {
        int idx = it * 2048 + tid * 8;          // element within 128x64 tile
        int sl  = idx >> 6;                     // local s: 0..127
        int d0  = idx & 63;                     // d, multiple of 8
        const float* src = vin + ((size_t)(b * S_LEN + s0 + sl)) * DMODEL + d0;
        v4f a = *(const v4f*)src;
        v4f c = *(const v4f*)(src + 4);
        int sp = (sl & ~31) | pv_perm(sl & 31); // permute within each 32-block
        #pragma unroll
        for (int e = 0; e < 4; ++e) {
            lt[d0 + e][sp]     = (_Float16)a[e];
            lt[d0 + 4 + e][sp] = (_Float16)c[e];
        }
    }
    __syncthreads();
    #pragma unroll
    for (int it = 0; it < 4; ++it) {
        int flat = it * 4096 + tid * 16;        // byte offset into 16KB tile
        int row  = flat >> 8;                   // d row (256B each)
        int off  = (flat & 255) >> 1;           // half offset in row
        u32x4 val = *(const u32x4*)(&lt[row][off]);
        *(u32x4*)(vt + ((size_t)(b * DMODEL + row)) * S_LEN + s0 + off) = val;
    }
}

// ---------------- Main flash-attention kernel ----------------
__global__ __launch_bounds__(256)
void fa_main(const float* __restrict__ q, const _Float16* __restrict__ kh,
             const _Float16* __restrict__ vt, float* __restrict__ o)
{
    __shared__ __align__(32) _Float16 kbuf[2][KV_TILE * DMODEL]; // [kv][d], 8KB each
    __shared__ __align__(32) _Float16 vbuf[2][DMODEL * KV_TILE]; // [d][perm kv], 8KB each
    __shared__ __align__(32) _Float16 pbuf[NWAVES][16 * KV_TILE]; // P, pair-packed

    const int lane = threadIdx.x & 31;
    const int wave = threadIdx.x >> 5;
    const int half = lane >> 4;
    const int l16  = lane & 15;

    const int b     = blockIdx.x >> 5;                 // 32 blocks per batch
    const int qbase = (blockIdx.x & 31) * QROWS_B + wave * QROWS_W;

    const _Float16* khb = kh + (size_t)b * S_LEN * DMODEL;
    const _Float16* vtb = vt + (size_t)b * DMODEL * S_LEN;

    // ---- Q tile as two A-operands (16x32 f16), scale folded in at load ----
    v16h qa[2];
    {
        const float* qrow = q + ((size_t)(b * S_LEN + qbase + l16)) * DMODEL;
        #pragma unroll
        for (int c = 0; c < 2; ++c) {
            const float* p0 = qrow + c * 32 + half * 16;
            const float* p1 = p0 + 8;
            #pragma unroll
            for (int i = 0; i < 8; ++i) {
                qa[c][i]     = (_Float16)(p0[i] * QSCALE);
                qa[c][8 + i] = (_Float16)(p1[i] * QSCALE);
            }
        }
    }

    // m: running row max (uniform across each 16-lane half after reduction).
    // lp: PER-LANE partial row sum; cross-lane reduced only once, after the loop.
    float m_r[8], lp[8];
    v8f acc[4];
    #pragma unroll
    for (int j = 0; j < 8; ++j) { m_r[j] = -1e30f; lp[j] = 0.0f; }
    #pragma unroll
    for (int g = 0; g < 4; ++g)
        #pragma unroll
        for (int j = 0; j < 8; ++j) acc[g][j] = 0.0f;

    // Cooperative stage of one KV tile: wave handles 8 rows of K and 8 rows of Vt.
    auto stage = [&](int buf, int kv0) {
        #pragma unroll
        for (int i = 0; i < 2; ++i) {
            int row = wave * 8 + i * 4 + (lane >> 3);  // 0..63 across block
            int seg = (lane & 7) * 8;                  // halves (16B segments)
            copy16(khb + (size_t)(kv0 + row) * DMODEL + seg, &kbuf[buf][row * DMODEL + seg]);
            copy16(vtb + (size_t)row * S_LEN + kv0 + seg, &vbuf[buf][row * KV_TILE + seg]);
        }
    };

    const int nT = S_LEN / KV_TILE;

    stage(0, 0);
    wait_async();
    __syncthreads();

    for (int t = 0; t < nT; ++t) {
        const int cur = t & 1;
        if (t + 1 < nT) stage(cur ^ 1, (t + 1) * KV_TILE);

        const _Float16* kb_ = kbuf[cur];
        const _Float16* vb_ = vbuf[cur];

        // ---- Scores (already in exp2 domain): 4 N=16 kv groups x 2 d-chunks ----
        v8f s[4];
        #pragma unroll
        for (int g = 0; g < 4; ++g)
            #pragma unroll
            for (int j = 0; j < 8; ++j) s[g][j] = 0.0f;

        #pragma unroll
        for (int g = 0; g < 4; ++g)
            #pragma unroll
            for (int c = 0; c < 2; ++c) {
                v16h kv = *(const v16h*)(kb_ + (g * 16 + l16) * DMODEL + c * 32 + half * 16);
                s[g] = __builtin_amdgcn_wmma_f32_16x16x32_f16(false, qa[c], false, kv,
                                                              (short)0, s[g], false, false);
            }

        // ---- Online softmax (exp2 domain; only the MAX is reduced per tile) ----
        float p[4][8];
        #pragma unroll
        for (int j = 0; j < 8; ++j) {
            float a0 = s[0][j], a1 = s[1][j], a2 = s[2][j], a3 = s[3][j];
            float rm = fmaxf(fmaxf(a0, a1), fmaxf(a2, a3));
            #pragma unroll
            for (int st = 0; st < 4; ++st)
                rm = fmaxf(rm, lane16_xor(rm, st));
            float mn    = fmaxf(m_r[j], rm);
            float alpha = fast_exp2(m_r[j] - mn);
            m_r[j] = mn;
            p[0][j] = fast_exp2(a0 - mn); p[1][j] = fast_exp2(a1 - mn);
            p[2][j] = fast_exp2(a2 - mn); p[3][j] = fast_exp2(a3 - mn);
            // Per-lane partial row sum: no cross-lane reduction inside the loop.
            lp[j] = lp[j] * alpha + ((p[0][j] + p[1][j]) + (p[2][j] + p[3][j]));
            #pragma unroll
            for (int g = 0; g < 4; ++g) acc[g][j] *= alpha;
        }

        // ---- P -> LDS as packed pairs (one b32 store per row per 32-block) ----
        _Float16* pl = pbuf[wave];
        #pragma unroll
        for (int j = 0; j < 8; ++j) {
            int r = j + 8 * half;
            v2h pk0; pk0[0] = (_Float16)p[0][j]; pk0[1] = (_Float16)p[1][j];
            v2h pk1; pk1[0] = (_Float16)p[2][j]; pk1[1] = (_Float16)p[3][j];
            *(v2h*)(pl + r * 32 + l16 * 2)       = pk0;   // kv block 0
            *(v2h*)(pl + 512 + r * 32 + l16 * 2) = pk1;   // kv block 1
        }
        asm volatile("s_wait_dscnt 0" ::: "memory");       // same-wave LDS RAW

        v16h pa[2];
        #pragma unroll
        for (int u = 0; u < 2; ++u)
            pa[u] = *(const v16h*)(pl + u * 512 + l16 * 32 + half * 16);

        // ---- O += P * V (perm-consistent contraction order) ----
        #pragma unroll
        for (int g = 0; g < 4; ++g)
            #pragma unroll
            for (int u = 0; u < 2; ++u) {
                v16h vv = *(const v16h*)(vb_ + (g * 16 + l16) * KV_TILE + u * 32 + half * 16);
                acc[g] = __builtin_amdgcn_wmma_f32_16x16x32_f16(false, pa[u], false, vv,
                                                                (short)0, acc[g], false, false);
            }

        wait_async();        // prefetched tile t+1 has landed
        __syncthreads();     // everyone done reading buf[cur]
    }

    // ---- Final cross-lane row-sum reduction (once), normalize, store ----
    float* obat = o + ((size_t)(b * S_LEN + qbase)) * DMODEL;
    #pragma unroll
    for (int j = 0; j < 8; ++j) {
        float rs = lp[j];
        #pragma unroll
        for (int st = 0; st < 4; ++st)
            rs += lane16_xor(rs, st);
        int row   = j + 8 * half;
        float inv = 1.0f / rs;
        #pragma unroll
        for (int g = 0; g < 4; ++g)
            obat[(size_t)row * DMODEL + g * 16 + l16] = acc[g][j] * inv;
    }
}

extern "C" void kernel_launch(void* const* d_in, const int* in_sizes, int n_in,
                              void* d_out, int out_size, void* d_ws, size_t ws_size,
                              hipStream_t stream) {
    const float* q = (const float*)d_in[0];
    const float* k = (const float*)d_in[1];
    const float* v = (const float*)d_in[2];
    float* o = (float*)d_out;

    const int B = in_sizes[0] / (S_LEN * DMODEL);          // 8
    _Float16* kh = (_Float16*)d_ws;                        // 4 MB
    _Float16* vt = kh + (size_t)B * S_LEN * DMODEL;        // 4 MB

    conv_k<<<dim3((B * S_LEN * DMODEL) / (8 * 256)), 256, 0, stream>>>(k, kh);
    conv_v<<<dim3(B * 32), 256, 0, stream>>>(v, vt);
    fa_main<<<dim3(B * 32), 256, 0, stream>>>(q, kh, vt, o);
}
